// HyperRHNCell_64879775973862
// MI455X (gfx1250) — matrix operand order repaired and checked by reading, not compiled, not run on hardware
//
#include <hip/hip_runtime.h>
#include <hip/hip_bf16.h>

// ---- problem constants (match reference) ----
#define B_TOT   32768
#define E_DIM   256
#define HH_DIM  128
#define HN_DIM  256
#define DEPTH_  5

// ---- tiling ----
#define MROWS    64      // rows of B per workgroup (4 row-tiles of 16)
#define KC       32      // K-chunk per TDM transfer == one WMMA k-step
#define SA       264     // LDS stride (elems) for x / state tiles (256 + pad)
#define SSH      136     // LDS stride for sH tile (128 + pad)
#define SB       40      // LDS stride for weight chunk rows: 64B data + 16B pad (TDM pad)
#define NTHREADS 256     // 8 wave32s

typedef __attribute__((ext_vector_type(16))) __bf16 v16bf;
typedef __attribute__((ext_vector_type(8)))  float  v8f;
typedef __attribute__((ext_vector_type(4)))  unsigned int ui32x4;
typedef __attribute__((ext_vector_type(8)))  int i32x8;
typedef __attribute__((ext_vector_type(4)))  int i32x4;

#define HAVE_TDM __has_builtin(__builtin_amdgcn_tensor_load_to_lds)

union FragU { v16bf v; unsigned int u[8]; };

// ---- bf16-weight workspace layout (elements), written by prepass ----
// transposed: dst[c][n][k] = (bf16) W[c][k][n]
#define WS_WIH 0                           // [256][256]
#define WS_WCH (WS_WIH + 256 * 256)        // [5][256][128]
#define WS_WUP (WS_WCH + 5 * 256 * 128)    // [5][256][128]
#define WS_WIN (WS_WUP + 5 * 256 * 128)    // [512][256]
#define WS_WCN (WS_WIN + 512 * 256)        // [5][512][256]
#define WS_TOTAL (WS_WCN + 5 * 512 * 256)  // 1,179,648 elems = 2.25 MiB bf16

// ---------------------------------------------------------------------------
// prepass: f32 [count][K][N] row-major  ->  bf16 [count][N][K] (transposed)
// ---------------------------------------------------------------------------
__global__ void convert_weights_kernel(const float* __restrict__ W,
                                       __bf16* __restrict__ dst,
                                       int K, int N, int count) {
  const size_t total = (size_t)count * K * N;
  for (size_t idx = (size_t)blockIdx.x * blockDim.x + threadIdx.x; idx < total;
       idx += (size_t)gridDim.x * blockDim.x) {
    const int n = (int)(idx % N);                 // fast index -> coalesced read
    const int k = (int)((idx / N) % K);
    const int c = (int)(idx / ((size_t)N * K));
    dst[((size_t)c * N + n) * K + k] = (__bf16)W[idx];
  }
}

// ---------------------------------------------------------------------------
// fragment loads (CDNA5 16-bit WMMA VGPR layouts, ISA 7.12.2)
// ---------------------------------------------------------------------------
__device__ __forceinline__ v16bf load_a_frag(const __bf16* lds, int stride,
                                             int m0, int k0, int lane) {
  const int kb = (lane & 16) ? 8 : 0;
  const int m  = m0 + (lane & 15);
  const unsigned int* row = (const unsigned int*)(lds + m * stride);
  FragU f;
#pragma unroll
  for (int j = 0; j < 8; ++j) {
    const int k = k0 + ((j & 4) ? 16 : 0) + kb + 2 * (j & 3);
    f.u[j] = row[k >> 1];
  }
  return f.v;
}

__device__ __forceinline__ v16bf load_b_frag(const __bf16* ldsB, int n0, int lane) {
  const int kb = (lane & 16) ? 8 : 0;    // lanes 0-15: K=0..7,16..23, lanes 16-31: K=8..15,24..31
  const int n  = n0 + (lane & 15);
  const unsigned int* col = (const unsigned int*)(ldsB + n * SB);
  FragU f;
#pragma unroll
  for (int j = 0; j < 8; ++j) f.u[j] = col[(kb + ((j & 4) ? 16 : 0) + 2 * (j & 3)) >> 1];
  return f.v;
}

// ---------------------------------------------------------------------------
// TDM: copy one [NW rows][KC bf16] chunk of transposed weights into LDS,
// inserting 16B padding after every 64B row (LDS pitch = SB*2 = 80B).
// ---------------------------------------------------------------------------
__device__ __forceinline__ void wait_weights() {
#if HAVE_TDM
  #if __has_builtin(__builtin_amdgcn_s_wait_tensorcnt)
  __builtin_amdgcn_s_wait_tensorcnt(0);
  #else
  asm volatile("s_wait_tensorcnt 0x0" ::: "memory");
  #endif
#endif
}

__device__ __forceinline__ void stage_chunk(const __bf16* __restrict__ Wt, int K,
                                            int kc, __bf16* bufB, int NW,
                                            int wave, int tid) {
#if HAVE_TDM
  if (wave == 0) {
    const unsigned long long ga = (unsigned long long)(Wt + kc);
    ui32x4 g0;
    g0[0] = 1u;                                            // count=1, user mode
    g0[1] = (unsigned)(unsigned long long)(const void*)bufB;  // lds_addr
    g0[2] = (unsigned)(ga & 0xffffffffu);                  // global_addr[31:0]
    g0[3] = (unsigned)((ga >> 32) & 0x01ffffffu) | 0x80000000u; // addr hi | type=2
    i32x8 g1;
    g1[0] = (1 << 16)      // data_size = 2 bytes
          | (1 << 20)      // pad_enable
          | (3 << 22)      // pad_interval: 16 DWORDs (one 64B row)
          | (3 << 25);     // pad_amount: 4 DWORDs (16B)
    g1[1] = (int)((K & 0xffff) << 16);                     // tensor_dim0 lo
    g1[2] = (int)(((unsigned)K >> 16) | ((unsigned)(NW & 0xffff) << 16)); // dim0 hi|dim1 lo
    g1[3] = (int)((((unsigned)NW >> 16) & 0xffffu) | ((unsigned)KC << 16)); // dim1 hi|tile_dim0
    g1[4] = (int)(NW & 0xffff);                            // tile_dim1 (tile_dim2=0)
    g1[5] = (int)K;                                        // tensor_dim0_stride lo
    g1[6] = 0;                                             // stride hi | dim1_stride lo
    g1[7] = 0;
    i32x4 z4; z4[0] = 0; z4[1] = 0; z4[2] = 0; z4[3] = 0;
    i32x8 z8; z8[0] = 0; z8[1] = 0; z8[2] = 0; z8[3] = 0;
    z8[4] = 0; z8[5] = 0; z8[6] = 0; z8[7] = 0;
    __builtin_amdgcn_tensor_load_to_lds(g0, g1, z4, z4, z8, 0);
  }
#else
  // fallback: cooperative direct copy (bf16 already transposed in ws)
  const int quads = NW * (KC / 4);
  for (int q = tid; q < quads; q += NTHREADS) {
    const int n  = q >> 3;                     // KC/4 == 8
    const int qq = q & 7;
    const unsigned long long v =
        *(const unsigned long long*)(Wt + (size_t)n * K + kc + qq * 4);
    *(unsigned long long*)(bufB + n * SB + qq * 4) = v;
  }
#endif
}

// ---------------------------------------------------------------------------
// K-chunked GEMM with TDM double buffering.
// acc[NT] += A(lds, bf16) * Wt(global bf16, transposed [NW][K]).
// Tile t covers cols nbase + (t/TG)*GOFF + (t%TG)*16.
// ---------------------------------------------------------------------------
template <int NT, int TG, int GOFF, int NW>
__device__ __forceinline__ void gemm_stage(v8f* acc, const __bf16* ldsA, int strideA,
                                           const __bf16* __restrict__ Wt, int K,
                                           __bf16* bufB0, __bf16* bufB1,
                                           int nbase, int m0, int lane, int wave,
                                           int tid) {
  stage_chunk(Wt, K, 0, bufB0, NW, wave, tid);
  for (int kc = 0; kc < K; kc += KC) {
    __bf16* cur = ((kc / KC) & 1) ? bufB1 : bufB0;
    __bf16* nxt = ((kc / KC) & 1) ? bufB0 : bufB1;
    if (wave == 0) wait_weights();
    __syncthreads();                       // chunk ready; prev readers of nxt done
    if (kc + KC < K) stage_chunk(Wt, K, kc + KC, nxt, NW, wave, tid);
    const v16bf a = load_a_frag(ldsA, strideA, m0, kc, lane);
#pragma unroll
    for (int t = 0; t < NT; ++t) {
      const int n0 = nbase + (t / TG) * GOFF + (t % TG) * 16;
      const v16bf b = load_b_frag(cur, n0, lane);
      acc[t] = __builtin_amdgcn_wmma_f32_16x16x32_bf16(
          false, a, false, b, (short)0, acc[t], false, false);
    }
  }
  __syncthreads();
}

// ---------------------------------------------------------------------------
// stage a [MROWS][RE] f32 tile -> bf16 LDS (float4 loads, packed b64 stores)
// ---------------------------------------------------------------------------
template <int RE>
__device__ __forceinline__ void stage_tile(const float* __restrict__ src,
                                           __bf16* dst, int strideDst, int tid) {
  const int qpr = RE / 4;
  for (int idx = tid; idx < MROWS * qpr; idx += NTHREADS) {
    const int m = idx / qpr;
    const int q = idx - m * qpr;
    const float4 f = ((const float4*)src)[(size_t)m * qpr + q];
    union { __bf16 h[4]; unsigned long long u; } p;
    p.h[0] = (__bf16)f.x; p.h[1] = (__bf16)f.y;
    p.h[2] = (__bf16)f.z; p.h[3] = (__bf16)f.w;
    *(unsigned long long*)(dst + m * strideDst + q * 4) = p.u;
  }
}

// ---------------------------------------------------------------------------
__global__ void __launch_bounds__(NTHREADS, 1)
hyperrhn_wmma_kernel(const float* __restrict__ x,
                     const float* __restrict__ sHyper,
                     const float* __restrict__ sNetwork,
                     const float* __restrict__ b_inp_h,
                     const float* __restrict__ b_cell_h,
                     const float* __restrict__ b_inp_n,
                     const float* __restrict__ b_cell_n,
                     const float* __restrict__ b_up,
                     const __bf16* __restrict__ ws,
                     float* __restrict__ out) {
  extern __shared__ char smem[];
  __bf16* bufX  = (__bf16*)smem;            // [64][SA]  resident x tile
  __bf16* bufA  = bufX + MROWS * SA;        // [64][SA]  current state tile
  __bf16* bufSH = bufA + MROWS * SA;        // [64][SSH] sH tile (A for z-GEMM)
  __bf16* bufB0 = bufSH + MROWS * SSH;      // [512][SB] weight chunk, buffer 0
  __bf16* bufB1 = bufB0 + 512 * SB;         // [512][SB] weight chunk, buffer 1

  const int tid  = threadIdx.x;
  const int lane = tid & 31;
  const int wave = tid >> 5;
  const int rt   = wave >> 1;   // row tile 0..3
  const int ch   = wave & 1;    // column half
  const int m0   = rt * 16;
  const int rowBase = blockIdx.x * MROWS;

  float* outMain = out;                                     // [B, 256]
  float* outSH   = out + (size_t)B_TOT * HN_DIM;            // [5, B, 128]
  float* outSN   = outSH + (size_t)DEPTH_ * B_TOT * HH_DIM; // [5, B, 256]

  stage_tile<E_DIM>(x + (size_t)rowBase * E_DIM, bufX, SA, tid);

  const int nlane = lane & 15;
  const int mlo   = (lane & 16) ? 8 : 0;

  for (int l = 0; l < DEPTH_; ++l) {
    // ---------------- Phase H: hyper RHN cell ----------------
    stage_tile<HH_DIM>(sHyper + ((size_t)l * B_TOT + rowBase) * HH_DIM, bufA, SA, tid);
    __syncthreads();

    v8f ah[8];
#pragma unroll
    for (int t = 0; t < 8; ++t)
#pragma unroll
      for (int v = 0; v < 8; ++v) ah[t][v] = 0.f;

    if (l == 0)
      gemm_stage<8, 4, 128, 256>(ah, bufX, SA, ws + WS_WIH, E_DIM,
                                 bufB0, bufB1, ch * 64, m0, lane, wave, tid);
    gemm_stage<8, 4, 128, 256>(ah, bufA, SA, ws + WS_WCH + (size_t)l * 256 * 128,
                               HH_DIM, bufB0, bufB1, ch * 64, m0, lane, wave, tid);

    // highway gate; write sH to global + LDS (A operand for z-GEMM)
#pragma unroll
    for (int t = 0; t < 4; ++t) {
      const int nc = ch * 64 + 16 * t + nlane;
      const int ng = nc + HH_DIM;
      float bc = b_cell_h[l * 2 * HH_DIM + nc];
      float bg = b_cell_h[l * 2 * HH_DIM + ng];
      if (l == 0) { bc += b_inp_h[nc]; bg += b_inp_h[ng]; }
#pragma unroll
      for (int v = 0; v < 8; ++v) {
        const int m  = m0 + mlo + v;
        const float wc = ah[t][v] + bc;
        const float wg = ah[t + 4][v] + bg;
        const float hh = tanhf(wc);
        const float tt = 1.f / (1.f + expf(-wg));
        const float s  = (float)bufA[m * SA + nc];
        const float o  = hh * tt + s * (1.f - tt);
        outSH[((size_t)l * B_TOT + rowBase + m) * HH_DIM + nc] = o;
        bufSH[m * SSH + nc] = (__bf16)o;
      }
    }
    __syncthreads();

    // ---------------- Phase Z: z = sH @ W_up + b_up (kept in registers) -----
    v8f zacc[8];
#pragma unroll
    for (int t = 0; t < 8; ++t)
#pragma unroll
      for (int v = 0; v < 8; ++v) zacc[t][v] = 0.f;

    gemm_stage<8, 8, 0, 256>(zacc, bufSH, SSH, ws + WS_WUP + (size_t)l * 256 * 128,
                             HH_DIM, bufB0, bufB1, ch * 128, m0, lane, wave, tid);
#pragma unroll
    for (int t = 0; t < 8; ++t) {
      const int nz = ch * 128 + 16 * t + nlane;
      const float bz = b_up[l * HN_DIM + nz];
#pragma unroll
      for (int v = 0; v < 8; ++v) zacc[t][v] += bz;
    }

    // ---------------- Phase N: main RHN cell ----------------
    stage_tile<HN_DIM>(sNetwork + ((size_t)l * B_TOT + rowBase) * HN_DIM, bufA, SA, tid);
    __syncthreads();

    v8f cn[16];
#pragma unroll
    for (int t = 0; t < 16; ++t)
#pragma unroll
      for (int v = 0; v < 8; ++v) cn[t][v] = 0.f;

    if (l == 0)
      gemm_stage<16, 8, 256, 512>(cn, bufX, SA, ws + WS_WIN, E_DIM,
                                  bufB0, bufB1, ch * 128, m0, lane, wave, tid);
    gemm_stage<16, 8, 256, 512>(cn, bufA, SA, ws + WS_WCN + (size_t)l * 512 * 256,
                                HN_DIM, bufB0, bufB1, ch * 128, m0, lane, wave, tid);

#pragma unroll
    for (int t = 0; t < 8; ++t) {
      const int nc = ch * 128 + 16 * t + nlane;
      const int ng = nc + HN_DIM;
      float bcn = b_cell_n[l * 2 * HN_DIM + nc];
      float bgn = b_cell_n[l * 2 * HN_DIM + ng];
      if (l == 0) { bcn += b_inp_n[nc]; bgn += b_inp_n[ng]; }
#pragma unroll
      for (int v = 0; v < 8; ++v) {
        const int m  = m0 + mlo + v;
        const float zv = zacc[t][v];
        const float wc = zv * (cn[t][v] + bcn);
        const float wg = zv * (cn[t + 8][v] + bgn);
        const float hh = tanhf(wc);
        const float tt = 1.f / (1.f + expf(-wg));
        const float s  = (float)bufA[m * SA + nc];
        const float o  = hh * tt + s * (1.f - tt);
        outSN[((size_t)l * B_TOT + rowBase + m) * HN_DIM + nc] = o;
        if (l == DEPTH_ - 1)
          outMain[(size_t)(rowBase + m) * HN_DIM + nc] = o;
      }
    }
    __syncthreads();
  }
}

extern "C" void kernel_launch(void* const* d_in, const int* in_sizes, int n_in,
                              void* d_out, int out_size, void* d_ws, size_t ws_size,
                              hipStream_t stream) {
  const float* x        = (const float*)d_in[0];
  const float* sHyper   = (const float*)d_in[1];
  const float* sNetwork = (const float*)d_in[2];
  const float* W_inp_h  = (const float*)d_in[3];
  const float* b_inp_h  = (const float*)d_in[4];
  const float* W_cell_h = (const float*)d_in[5];
  const float* b_cell_h = (const float*)d_in[6];
  const float* W_inp_n  = (const float*)d_in[7];
  const float* b_inp_n  = (const float*)d_in[8];
  const float* W_cell_n = (const float*)d_in[9];
  const float* b_cell_n = (const float*)d_in[10];
  const float* W_up     = (const float*)d_in[11];
  const float* b_up     = (const float*)d_in[12];
  float* out  = (float*)d_out;
  __bf16* ws  = (__bf16*)d_ws;   // needs WS_TOTAL*2 = 2,359,296 bytes

  // prepass: convert + transpose all weights to bf16 in workspace
  convert_weights_kernel<<<512, NTHREADS, 0, stream>>>(W_inp_h,  ws + WS_WIH, 256, 256, 1);
  convert_weights_kernel<<<512, NTHREADS, 0, stream>>>(W_cell_h, ws + WS_WCH, 128, 256, 5);
  convert_weights_kernel<<<512, NTHREADS, 0, stream>>>(W_up,     ws + WS_WUP, 128, 256, 5);
  convert_weights_kernel<<<512, NTHREADS, 0, stream>>>(W_inp_n,  ws + WS_WIN, 256, 512, 1);
  convert_weights_kernel<<<512, NTHREADS, 0, stream>>>(W_cell_n, ws + WS_WCN, 256, 512, 5);

  const size_t smemBytes =
      (size_t)(MROWS * SA + MROWS * SA + MROWS * SSH + 2 * 512 * SB) * sizeof(__bf16);

  (void)hipFuncSetAttribute((const void*)hyperrhn_wmma_kernel,
                            hipFuncAttributeMaxDynamicSharedMemorySize,
                            (int)smemBytes);

  hyperrhn_wmma_kernel<<<dim3(B_TOT / MROWS), dim3(NTHREADS), smemBytes, stream>>>(
      x, sHyper, sNetwork, b_inp_h, b_cell_h, b_inp_n, b_cell_n, b_up, ws, out);
}
